// WinRelMultiHeadedSelfAttention_26079041421958
// MI455X (gfx1250) — compile-verified, matrix operand order checked by reading, not run on hardware
//
#include <hip/hip_runtime.h>

#define BB   8
#define TT   512
#define DDIM 1024
#define HH   16
#define DK   64
#define SS   1023

typedef __attribute__((ext_vector_type(8)))  __bf16 v8bf;
typedef __attribute__((ext_vector_type(16))) __bf16 v16bf;
typedef __attribute__((ext_vector_type(8)))  float  v8f;

static __device__ __forceinline__ unsigned short f2bf(float f) {
  unsigned int u = __float_as_uint(f);
  u += 0x7FFFu + ((u >> 16) & 1u);           // round-to-nearest-even
  return (unsigned short)(u >> 16);
}
static __device__ __forceinline__ float bf2f(unsigned short s) {
  return __uint_as_float(((unsigned int)s) << 16);
}

// Load one 16-element bf16 WMMA operand fragment for this lane.
// ISA 16-bit A/B layout: lane holds K = [hi*8 .. hi*8+7] and [16+hi*8 .. 16+hi*8+7]
// (caller passes p already offset by row*ld + kbase + hi*8).
static __device__ __forceinline__ v16bf ld_frag(const unsigned short* p) {
  v8bf lo = *reinterpret_cast<const v8bf*>(p);
  v8bf hp = *reinterpret_cast<const v8bf*>(p + 16);
  return __builtin_shufflevector(lo, hp, 0,1,2,3,4,5,6,7,8,9,10,11,12,13,14,15);
}
static __device__ __forceinline__ v8f wmma_bf16(v16bf a, v16bf b, v8f c) {
  return __builtin_amdgcn_wmma_f32_16x16x32_bf16(false, a, false, b, (short)0, c, false, false);
}

// Async 16-byte global -> LDS copy (tracked by ASYNCcnt).
// LDS dest address = low 32 bits of the generic pointer (LDS aperture maps
// addr[31:0] directly to the wave-relative LDS byte address).
static __device__ __forceinline__ void async_ld_b128(const unsigned short* gsrc,
                                                     unsigned short* ldst) {
  unsigned laddr = (unsigned)(uintptr_t)ldst;
  asm volatile("global_load_async_to_lds_b128 %0, %1, off"
               :: "v"(laddr), "v"(gsrc)
               : "memory");
}
static __device__ __forceinline__ void wait_async0() {
  asm volatile("s_wait_asynccnt 0x0" ::: "memory");
}

// ---------------------------------------------------------------------------
// fp32 -> bf16 convert
// ---------------------------------------------------------------------------
__global__ __launch_bounds__(256) void cvt_f32_bf16(
    const float* __restrict__ in, unsigned short* __restrict__ out, int n) {
  int i = blockIdx.x * 256 + threadIdx.x;
  if (i < n) out[i] = f2bf(in[i]);
}

// ---------------------------------------------------------------------------
// Unified bf16 GEMM, C[m,n] = sum_k A[m,k]*Bm[n,k], K = 1024.
// Block = 4 waves sharing one 16-row A tile (staged to LDS via async copies);
// each wave computes a 16x64 strip (A frag reused over 4 B tiles).
// Requires Ntiles64 % 4 == 0 and an exact grid (no remainder waves).
// mode 0: QKV scatter epilogue   mode 1: pos-projection scatter   mode 2: out
// ---------------------------------------------------------------------------
__global__ __launch_bounds__(128) void gemm_bf16_k1024(
    const unsigned short* __restrict__ A,
    const unsigned short* __restrict__ Bm,
    int M, int Ntiles64, int mode,
    const float* __restrict__ bias,
    const float* __restrict__ posu,
    const float* __restrict__ posv,
    unsigned short* __restrict__ o_qu,
    unsigned short* __restrict__ o_qv,
    unsigned short* __restrict__ o_k,
    unsigned short* __restrict__ o_vT,
    unsigned short* __restrict__ o_p,
    float* __restrict__ o_out)
{
  __shared__ __align__(16) unsigned short sA[16][1024];    // 32 KB A tile

  const int tid = threadIdx.x;
  const int mt  = (blockIdx.x * 4) / Ntiles64;             // same for all 4 waves

  // ---- cooperative async stage of A tile: 16 rows x 1024 bf16 (16B chunks) --
  for (int c = tid; c < 16 * 128; c += 128) {
    const int row  = c >> 7;
    const int col8 = (c & 127) << 3;
    int rA = mt * 16 + row;
    if (rA >= M) rA = M - 1;
    async_ld_b128(A + (size_t)rA * 1024 + col8, &sA[row][col8]);
  }
  wait_async0();
  __syncthreads();

  const int wid  = blockIdx.x * 4 + (tid >> 5);
  const int nt   = wid % Ntiles64;
  const int lane = tid & 31;
  const int ln   = lane & 15;
  const int hi   = lane >> 4;

  const unsigned short* b0 = Bm + (size_t)(nt * 64 +  0 + ln) * 1024 + hi * 8;
  const unsigned short* b1 = Bm + (size_t)(nt * 64 + 16 + ln) * 1024 + hi * 8;
  const unsigned short* b2 = Bm + (size_t)(nt * 64 + 32 + ln) * 1024 + hi * 8;
  const unsigned short* b3 = Bm + (size_t)(nt * 64 + 48 + ln) * 1024 + hi * 8;

  v8f acc0 = {}, acc1 = {}, acc2 = {}, acc3 = {};
  for (int kb = 0; kb < 1024; kb += 32) {
    __builtin_prefetch(b0 + kb + 256, 0, 0);               // global_prefetch_b8
    // hoist all fragment loads so VMEM issues overlap, then 4 WMMAs
    v16bf a  = ld_frag(&sA[ln][kb + hi * 8]);              // ds_load_b128 x2
    v16bf f0 = ld_frag(b0 + kb);
    v16bf f1 = ld_frag(b1 + kb);
    v16bf f2 = ld_frag(b2 + kb);
    v16bf f3 = ld_frag(b3 + kb);
    acc0 = wmma_bf16(a, f0, acc0);
    acc1 = wmma_bf16(a, f1, acc1);
    acc2 = wmma_bf16(a, f2, acc2);
    acc3 = wmma_bf16(a, f3, acc3);
  }

  v8f accs[4] = {acc0, acc1, acc2, acc3};
#pragma unroll
  for (int u = 0; u < 4; ++u) {
    const int n = nt * 64 + u * 16 + ln;
#pragma unroll
    for (int r = 0; r < 8; ++r) {
      const int m = mt * 16 + hi * 8 + r;
      if (m >= M) continue;
      float val = accs[u][r];
      if (mode == 0) {
        val += bias[n];
        const int b = m >> 9, t = m & 511;
        if (n < 1024) {                         // Q third
          const int h = n >> 6, dk = n & 63;
          const size_t off = (((size_t)(b * HH + h)) * TT + t) * DK + dk;
          o_qu[off] = f2bf(val + posu[n]);
          o_qv[off] = f2bf(val + posv[n]);
        } else if (n < 2048) {                  // K third
          const int n2 = n - 1024;
          const int h = n2 >> 6, dk = n2 & 63;
          o_k[(((size_t)(b * HH + h)) * TT + t) * DK + dk] = f2bf(val);
        } else {                                // V third, stored transposed [64][512]
          const int n2 = n - 2048;
          const int h = n2 >> 6, dk = n2 & 63;
          o_vT[(((size_t)(b * HH + h)) * DK + dk) * TT + t] = f2bf(val);
        }
      } else if (mode == 1) {                   // p[h][s][dk], no bias
        const int h = n >> 6, dk = n & 63;
        o_p[(((size_t)h) * SS + m) * DK + dk] = f2bf(val);
      } else {                                  // out = ctx@Wout^T + bout
        o_out[(size_t)m * 1024 + n] = val + bias[n];
      }
    }
  }
}

// ---------------------------------------------------------------------------
// Fused relative attention: one block per (b, h, 16-row query tile).
//  AC   = (q+posu) k^T            via WMMA -> LDS fp32 [16][512]
//  band = (q+posv) p^T restricted to the 527-wide diagonal band -> LDS bf16
//  scores = (AC + band[r][j+15-r]) * 1/8, masked, softmax (wave32 shfl)
//  weights stored fp32 to global; PV GEMM (split-K over 8 waves, ds_add)
// ---------------------------------------------------------------------------
__global__ __launch_bounds__(256) void attn_relpos(
    const unsigned short* __restrict__ qu,
    const unsigned short* __restrict__ qv,
    const unsigned short* __restrict__ kmat,
    const unsigned short* __restrict__ vT,
    const unsigned short* __restrict__ pmat,
    const unsigned char*  __restrict__ mask,
    float* __restrict__ weights,
    unsigned short* __restrict__ ctxb)
{
  __shared__ float sc[16][512];                       // 32 KB
  union ShU {
    unsigned short band[16][528];                     // 16.5 KB
    unsigned short wbf[16][512];                      // aliased after softmax
  };
  __shared__ __align__(32) ShU u;
  __shared__ float ctxs[16][64];                      // 4 KB

  const int blk = blockIdx.x;
  const int t0  = (blk & 31) << 4;
  const int bh  = blk >> 5;
  const int h   = bh & (HH - 1);
  const int b   = bh >> 4;
  const int tid  = threadIdx.x;
  const int lane = tid & 31;
  const int wave = tid >> 5;
  const int ln   = lane & 15;
  const int hi   = lane >> 4;

  const unsigned short* quB = qu   + (size_t)bh * TT * DK;
  const unsigned short* qvB = qv   + (size_t)bh * TT * DK;
  const unsigned short* kB  = kmat + (size_t)bh * TT * DK;
  const unsigned short* vB  = vT   + (size_t)bh * DK * TT;
  const unsigned short* pB  = pmat + (size_t)h  * SS * DK;

  // A fragments for this wave's row tile (rows t0..t0+15), K = 64 -> 2 frags
  v16bf aqu0, aqu1, aqv0, aqv1;
  {
    const unsigned short* r0 = quB + (size_t)(t0 + ln) * DK + hi * 8;
    aqu0 = ld_frag(r0); aqu1 = ld_frag(r0 + 32);
    const unsigned short* r1 = qvB + (size_t)(t0 + ln) * DK + hi * 8;
    aqv0 = ld_frag(r1); aqv1 = ld_frag(r1 + 32);
  }

  // -------- AC: 32 column tiles, 8 waves round-robin --------
  for (int jt = wave; jt < 32; jt += 8) {
    const unsigned short* br = kB + (size_t)(jt * 16 + ln) * DK + hi * 8;
    v16bf bb0 = ld_frag(br);
    v16bf bb1 = ld_frag(br + 32);
    v8f c = {};
    c = wmma_bf16(aqu0, bb0, c);
    c = wmma_bf16(aqu1, bb1, c);
#pragma unroll
    for (int r = 0; r < 8; ++r) sc[hi * 8 + r][jt * 16 + ln] = c[r];
  }

  // -------- BD band: s in [496-t0, 1022-t0], 33 tiles of 16 --------
  const int s_lo = 496 - t0;                          // always >= 0
  for (int ct = wave; ct < 33; ct += 8) {
    int srow = s_lo + ct * 16 + ln;
    if (srow > SS - 1) srow = SS - 1;                 // col 527 never gathered
    const unsigned short* br = pB + (size_t)srow * DK + hi * 8;
    v16bf bb0 = ld_frag(br);
    v16bf bb1 = ld_frag(br + 32);
    v8f c = {};
    c = wmma_bf16(aqv0, bb0, c);
    c = wmma_bf16(aqv1, bb1, c);
#pragma unroll
    for (int r = 0; r < 8; ++r) u.band[hi * 8 + r][ct * 16 + ln] = f2bf(c[r]);
  }
  __syncthreads();

  // -------- combine AC + gathered BD, scale, mask --------
  const unsigned char* mrow = mask + ((size_t)b * TT + t0) * TT;
  for (int idx = tid; idx < 16 * 512; idx += 256) {
    const int r = idx >> 9, j = idx & 511;
    const float v = (sc[r][j] + bf2f(u.band[r][j + 15 - r])) * 0.125f;
    sc[r][j] = mrow[(size_t)r * TT + j] ? v : -100000.0f;
  }
  __syncthreads();

  // -------- softmax: each wave owns rows {wave, wave+8} --------
#pragma unroll
  for (int rr = 0; rr < 2; ++rr) {
    const int r = wave + rr * 8;
    float mx = -3.0e38f;
    for (int j = lane; j < 512; j += 32) mx = fmaxf(mx, sc[r][j]);
#pragma unroll
    for (int o = 16; o > 0; o >>= 1) mx = fmaxf(mx, __shfl_xor(mx, o, 32));
    float sum = 0.f;
    for (int j = lane; j < 512; j += 32) {
      const float e = __expf(sc[r][j] - mx);
      sc[r][j] = e;
      sum += e;
    }
#pragma unroll
    for (int o = 16; o > 0; o >>= 1) sum += __shfl_xor(sum, o, 32);
    const float inv = 1.0f / sum;
    for (int j = lane; j < 512; j += 32) sc[r][j] *= inv;
  }
  __syncthreads();

  // -------- emit fp32 weights + bf16 copy for PV --------
  float* wout = weights + ((size_t)bh * TT + t0) * TT;
  for (int idx = tid; idx < 16 * 512; idx += 256) {
    const int r = idx >> 9, j = idx & 511;
    const float wv = sc[r][j];
    wout[(size_t)r * TT + j] = wv;
    u.wbf[r][j] = f2bf(wv);
  }
  for (int idx = tid; idx < 16 * 64; idx += 256) ((float*)ctxs)[idx] = 0.f;
  __syncthreads();

  // -------- PV: ctx[16][64] = W[16][512] x V[512][64], split-K over 8 waves --
  {
    const int dt = wave & 3;        // N tile (16 of 64 dk columns)
    const int kh = wave >> 2;       // K half
    const unsigned short* br = vB + (size_t)(dt * 16 + ln) * TT + hi * 8;
    v8f c = {};
    for (int ks = kh * 8; ks < kh * 8 + 8; ++ks) {
      v16bf a  = ld_frag(&u.wbf[ln][ks * 32 + hi * 8]);
      v16bf bb = ld_frag(br + ks * 32);
      c = wmma_bf16(a, bb, c);
    }
#pragma unroll
    for (int r = 0; r < 8; ++r)
      atomicAdd(&ctxs[hi * 8 + r][dt * 16 + ln], c[r]);
  }
  __syncthreads();

  // ctx layout [B,T,D] with d = h*64 + dk  (matches bthd reshape)
  for (int idx = tid; idx < 16 * 64; idx += 256) {
    const int r = idx >> 6, d = idx & 63;
    ctxb[((size_t)(b * TT) + t0 + r) * DDIM + h * DK + d] = f2bf(ctxs[r][d]);
  }
}

// ---------------------------------------------------------------------------
extern "C" void kernel_launch(void* const* d_in, const int* in_sizes, int n_in,
                              void* d_out, int out_size, void* d_ws, size_t ws_size,
                              hipStream_t stream) {
  (void)in_sizes; (void)n_in; (void)out_size; (void)ws_size;
  const float*         x    = (const float*)d_in[0];
  const unsigned char* mask = (const unsigned char*)d_in[1];
  const float*         pos  = (const float*)d_in[2];
  const float*         Wqkv = (const float*)d_in[3];
  const float*         bqkv = (const float*)d_in[4];
  const float*         Wpos = (const float*)d_in[5];
  const float*         posu = (const float*)d_in[6];
  const float*         posv = (const float*)d_in[7];
  const float*         Wout = (const float*)d_in[8];
  const float*         bout = (const float*)d_in[9];

  float* out     = (float*)d_out;
  float* weights = out + (size_t)BB * TT * DDIM;

  char* wp = (char*)d_ws;
  auto take = [&wp](size_t bytes) -> char* {
    char* p = wp;
    wp += (bytes + 255) & ~(size_t)255;
    return p;
  };
  unsigned short* xb    = (unsigned short*)take((size_t)4096 * 1024 * 2);
  unsigned short* wqkvb = (unsigned short*)take((size_t)3072 * 1024 * 2);
  unsigned short* wposb = (unsigned short*)take((size_t)1024 * 1024 * 2);
  unsigned short* posb  = (unsigned short*)take((size_t)SS * 1024 * 2);
  unsigned short* woutb = (unsigned short*)take((size_t)1024 * 1024 * 2);
  unsigned short* qub   = (unsigned short*)take((size_t)BB * HH * TT * DK * 2);
  unsigned short* qvb   = (unsigned short*)take((size_t)BB * HH * TT * DK * 2);
  unsigned short* kbuf  = (unsigned short*)take((size_t)BB * HH * TT * DK * 2);
  unsigned short* vTb   = (unsigned short*)take((size_t)BB * HH * TT * DK * 2);
  unsigned short* pbuf  = (unsigned short*)take((size_t)HH * SS * DK * 2);
  unsigned short* ctxb  = (unsigned short*)take((size_t)4096 * 1024 * 2);

  auto cvt = [&](const float* src, unsigned short* dst, int n) {
    cvt_f32_bf16<<<(n + 255) / 256, 256, 0, stream>>>(src, dst, n);
  };
  cvt(x,    xb,    4096 * 1024);
  cvt(Wqkv, wqkvb, 3072 * 1024);
  cvt(Wpos, wposb, 1024 * 1024);
  cvt(pos,  posb,  SS * 1024);
  cvt(Wout, woutb, 1024 * 1024);

  // QKV projection: [4096,1024] x [3072,1024]^T  (scatter q/k/v, fold posu/posv)
  {
    const int M = 4096, Nt = 48;
    const int waves = (M / 16) * Nt;                 // 12288
    gemm_bf16_k1024<<<waves / 4, 128, 0, stream>>>(
        xb, wqkvb, M, Nt, 0, bqkv, posu, posv,
        qub, qvb, kbuf, vTb, nullptr, nullptr);
  }
  // positional projection: [1023,1024] x [1024,1024]^T -> p[h][s][dk]
  {
    const int M = SS, Nt = 16;
    const int Mt = (M + 15) / 16;                    // 64
    gemm_bf16_k1024<<<(Mt * Nt) / 4, 128, 0, stream>>>(
        posb, wposb, M, Nt, 1, nullptr, nullptr, nullptr,
        nullptr, nullptr, nullptr, nullptr, pbuf, nullptr);
  }
  // fused attention (AC + banded BD + softmax + weights + PV)
  attn_relpos<<<BB * HH * (TT / 16), 256, 0, stream>>>(
      qub, qvb, kbuf, vTb, pbuf, mask, weights, ctxb);
  // output projection: [4096,1024] x [1024,1024]^T + bout
  {
    const int M = 4096, Nt = 16;
    const int waves = (M / 16) * Nt;                 // 4096
    gemm_bf16_k1024<<<waves / 4, 128, 0, stream>>>(
        ctxb, woutb, M, Nt, 2, bout, nullptr, nullptr,
        nullptr, nullptr, nullptr, nullptr, nullptr, out);
  }
}
